// WaveNetAR_77025943486501
// MI455X (gfx1250) — matrix operand order repaired and checked by reading, not compiled, not run on hardware
//
#include <hip/hip_runtime.h>

// ---------------------------------------------------------------------------
// Autoregressive WaveNet generation for MI455X (gfx1250, wave32, WMMA).
// Single persistent workgroup runs the 2048-step sequential scan; all GEMMs
// use v_wmma_f32_16x16x32_f16 (M=16 batch tile). Weights are pre-packed into
// WMMA B-fragment layout; dilation ring buffers AND the conditioner
// projection inputs are stored in WMMA A-fragment layout so every matrix
// operand loads directly into v16h registers with global_load_b128 pairs.
// ---------------------------------------------------------------------------

typedef __attribute__((ext_vector_type(16))) _Float16 v16h;
typedef __attribute__((ext_vector_type(8)))  _Float16 v8h;
typedef __attribute__((ext_vector_type(8)))  float    v8f;

#define T_STEPS 2048
#define B_SZ    16
#define R_SZ    128
#define CC_SZ   80
#define L_SZ    8

// Workspace layout (units: _Float16 halves)
#define WS_WC     0u         // conv W packed: 8l x 2tap x 16tile x 4kb x 512 = 524288
#define WS_WCOND  524288u    // cond W packed: 8l x 16tile x 3kb x 512        = 196608
#define WS_WRES   720896u    // res  W packed: 8l x 8tile x 4kb x 512         = 131072
#define WS_WOUT1  851968u    // out1 W packed: 8tile x 32kk x 512             = 131072
#define WS_RING   983040u    // rings: sum_l (2^l + 1) slots * 2048 halves    = 538624
#define WS_CONDA  1521664u   // cond input A-frags: 2048t x 3kb x 512         = 3145728
#define WS_BIASZ  4667392u   // fused conv+cond bias: 2048 floats (4096 halves)
#define RING_UINTS 269312u   // 538624/2

__device__ __forceinline__ v8f wmma16(v16h a, v16h b, v8f c) {
  return __builtin_amdgcn_wmma_f32_16x16x32_f16(false, a, false, b, (short)0, c,
                                                false, false);
}

// A-fragment element map (contiguous, matches B packing): lane holds row = lane&15;
// k0 = kbase + (lane>=16 ? 16 : 0); halves 0..15 -> K=k0..k0+15.
__device__ __forceinline__ v16h ldsAfrag(const _Float16* base, int stride,
                                         int kbase, int lane) {
  int row = lane & 15;
  int k0  = kbase + ((lane & 16) ? 16 : 0);
  const _Float16* p = base + row * stride + k0;
  v8h lo = *(const v8h*)(p);
  v8h hi = *(const v8h*)(p + 8);
  v16h r;
#pragma unroll
  for (int i = 0; i < 8; ++i) { r[i] = lo[i]; r[i + 8] = hi[i]; }
  return r;
}

// ---------------------------- prep kernels ---------------------------------
// B-fragment pack: lane n<16 -> N=n, K=kb*32+h ; lane>=16 -> N=n-16, K=kb*32+16+h.

__global__ void pack_conv_k(const float* __restrict__ W, _Float16* __restrict__ ws) {
  unsigned gid = blockIdx.x * blockDim.x + threadIdx.x;
  if (gid >= 524288u) return;
  unsigned h = gid & 15u, lane = (gid >> 4) & 31u, kb = (gid >> 9) & 3u,
           tile = (gid >> 11) & 15u, tp = (gid >> 15) & 1u, l = gid >> 16;
  unsigned K = kb * 32u + ((lane >> 4) & 1u) * 16u + h;
  unsigned N = tile * 16u + (lane & 15u);
  ws[WS_WC + gid] = (_Float16)W[((l * 256u + N) * 128u + K) * 2u + tp];
}

__global__ void pack_cond_k(const float* __restrict__ W, _Float16* __restrict__ ws) {
  unsigned gid = blockIdx.x * blockDim.x + threadIdx.x;
  if (gid >= 196608u) return;
  unsigned h = gid & 15u, lane = (gid >> 4) & 31u, r = gid >> 9;
  unsigned kb = r % 3u, rem = r / 3u, tile = rem & 15u, l = rem >> 4;
  unsigned K = kb * 32u + ((lane >> 4) & 1u) * 16u + h;   // K padded 80 -> 96
  unsigned N = tile * 16u + (lane & 15u);
  float v = (K < CC_SZ) ? W[(l * 256u + N) * CC_SZ + K] : 0.f;
  ws[WS_WCOND + gid] = (_Float16)v;
}

__global__ void pack_res_k(const float* __restrict__ W, _Float16* __restrict__ ws) {
  unsigned gid = blockIdx.x * blockDim.x + threadIdx.x;
  if (gid >= 131072u) return;
  unsigned h = gid & 15u, lane = (gid >> 4) & 31u, kb = (gid >> 9) & 3u,
           tile = (gid >> 11) & 7u, l = gid >> 14;
  unsigned K = kb * 32u + ((lane >> 4) & 1u) * 16u + h;
  unsigned N = tile * 16u + (lane & 15u);
  ws[WS_WRES + gid] = (_Float16)W[(l * 128u + N) * 128u + K];
}

__global__ void pack_out1_k(const float* __restrict__ W, _Float16* __restrict__ ws) {
  unsigned gid = blockIdx.x * blockDim.x + threadIdx.x;
  if (gid >= 131072u) return;
  unsigned h = gid & 15u, lane = (gid >> 4) & 31u, kk = (gid >> 9) & 31u,
           tile = gid >> 14;
  unsigned K = kk * 32u + ((lane >> 4) & 1u) * 16u + h;   // K over 1024 = 8l*128
  unsigned N = tile * 16u + (lane & 15u);
  ws[WS_WOUT1 + gid] = (_Float16)W[N * 1024u + K];
}

// Pre-transpose cond_input (B,CCOND,T) into per-step A-fragments (K padded to 96):
// frag[t][kb][lane][h] = cond[row = lane&15][K][t], fully parallel over t.
__global__ void pack_condin_k(const float* __restrict__ C, _Float16* __restrict__ ws) {
  unsigned gid = blockIdx.x * blockDim.x + threadIdx.x;
  if (gid >= 3145728u) return;
  unsigned h = gid & 15u, lane = (gid >> 4) & 31u, kb = (gid >> 9) % 3u,
           t = (gid >> 9) / 3u;
  unsigned row = lane & 15u;
  unsigned K = kb * 32u + ((lane >> 4) & 1u) * 16u + h;
  float v = (K < CC_SZ) ? C[(row * CC_SZ + K) * T_STEPS + t] : 0.f;
  ws[WS_CONDA + gid] = (_Float16)v;
}

__global__ void pack_biasz_k(const float* __restrict__ bconv,
                             const float* __restrict__ bcond,
                             _Float16* __restrict__ ws) {
  unsigned gid = blockIdx.x * blockDim.x + threadIdx.x;
  if (gid < 2048u) ((float*)(ws + WS_BIASZ))[gid] = bconv[gid] + bcond[gid];
}

__global__ void zero_ring_k(_Float16* __restrict__ ws) {
  unsigned gid = blockIdx.x * blockDim.x + threadIdx.x;
  if (gid < RING_UINTS) ((unsigned*)(ws + WS_RING))[gid] = 0u;
}

// ------------------------- persistent AR kernel ----------------------------

__launch_bounds__(512, 1)
__global__ void wavenet_ar_k(const float* __restrict__ W_in,
                             const float* __restrict__ b_in,
                             const float* __restrict__ b_res,
                             const float* __restrict__ b_out1,
                             const float* __restrict__ W_out2,
                             const float* __restrict__ b_out2,
                             _Float16* __restrict__ ws,
                             float* __restrict__ out) {
  // LDS (bytes):
  //   [0,4352)      xnat : f16 [16][136]    layer input x
  //   [4352,39168)  snat : f16 [8][16][136] gated activations (skips)
  //   [39168,55552) zt   : f32 [16][32][8]  (reused as hnat f32[16][136])
  //   [55552,55616) ybuf : f32 [16]
  __shared__ __align__(32) unsigned char smem[55616];
  _Float16* xnat = (_Float16*)(smem);
  _Float16* snat = (_Float16*)(smem + 4352);
  float*    zt   = (float*)(smem + 39168);
  float*    hnat = (float*)(smem + 39168);
  float*    ybuf = (float*)(smem + 55552);

  const int tid   = threadIdx.x;
  const int w     = tid >> 5;            // wave 0..15
  const int lane  = tid & 31;
  const int col16 = lane & 15;
  const int mb    = (lane & 16) ? 8 : 0; // C/D row base for this lane half

  const _Float16* wc    = ws + WS_WC;
  const _Float16* wcond = ws + WS_WCOND;
  const _Float16* wres  = ws + WS_WRES;
  const _Float16* wout1 = ws + WS_WOUT1;
  const _Float16* conda = ws + WS_CONDA;
  const float*    biasz = (const float*)(ws + WS_BIASZ);
  _Float16*       ring  = ws + WS_RING;

#pragma unroll 1
  for (int t = 0; t < T_STEPS; ++t) {
    // ---- phase 0: x = tanh(y_prev * W_in + b_in)
    for (int i = tid; i < 16 * 128; i += 512) {
      int m = i >> 7, r = i & 127;
      float yp = (t == 0) ? 0.f : ybuf[m];
      xnat[m * 136 + r] = (_Float16)tanhf(yp * W_in[r] + b_in[r]);
    }
    __syncthreads();

#pragma unroll 1
    for (int l = 0; l < L_SZ; ++l) {
      const int d     = 1 << l;
      const int slots = d + 1;            // extra slot: no read/write hazard
      const int rslot = (t + 1) % slots;  // == (t - d) mod (d+1)
      const int wslot = t % slots;
      _Float16* ringl = ring + (unsigned)((d - 1) + l) * 2048u;

      // ---- phase 1 (16 waves): z tile w = tap*Wc0 + x*Wc1 + cond*Wcond + bias
      {
        float bz = biasz[l * 256 + w * 16 + col16];
        v8f acc;
#pragma unroll
        for (int v = 0; v < 8; ++v) acc[v] = bz;

        const _Float16* bp0 = wc + (unsigned)((((l * 2 + 0) * 16 + w) * 4) * 512) + lane * 16;
        const _Float16* bp1 = wc + (unsigned)((((l * 2 + 1) * 16 + w) * 4) * 512) + lane * 16;
        if (l < 7)  // warm next layer's weights while computing this one
          __builtin_prefetch((const void*)(wc + (unsigned)(((((l + 1) * 2) * 16 + w) * 4) * 512) + lane * 16), 0, 0);

#pragma unroll
        for (int kb = 0; kb < 4; ++kb) {   // dilated tap (A-frags from ring)
          v16h aT = *(const v16h*)(ringl + (unsigned)rslot * 2048u + kb * 512 + lane * 16);
          acc = wmma16(aT, *(const v16h*)(bp0 + kb * 512), acc);
        }
#pragma unroll
        for (int kb = 0; kb < 4; ++kb) {   // current x (A-frags gathered from LDS)
          v16h aX = ldsAfrag(xnat, 136, kb * 32, lane);
          acc = wmma16(aX, *(const v16h*)(bp1 + kb * 512), acc);
        }
        const _Float16* bpc = wcond + (unsigned)(((l * 16 + w) * 3) * 512) + lane * 16;
        const _Float16* apc = conda + (unsigned)(t * 3) * 512u + lane * 16;
#pragma unroll
        for (int kb = 0; kb < 3; ++kb) {   // conditioner (pre-built A-frags)
          v16h aC = *(const v16h*)(apc + kb * 512);
          acc = wmma16(aC, *(const v16h*)(bpc + kb * 512), acc);
        }
        *(v8f*)(zt + (w * 32 + lane) * 8) = acc;

        // push current layer input x into this layer's dilation ring (A-frag layout)
        if (w < 4) {
          v16h xf = ldsAfrag(xnat, 136, w * 32, lane);
          *(v16h*)(ringl + (unsigned)wslot * 2048u + w * 512 + lane * 16) = xf;
        }
      }
      asm volatile("s_wait_storecnt 0x0" ::: "memory");  // ring push visible
      __syncthreads();

      // ---- phase 2a (waves 0..7): a = tanh(z_lo) * sigmoid(z_hi)
      if (w < 8) {
        v8f zlo = *(const v8f*)(zt + (w * 32 + lane) * 8);
        v8f zhi = *(const v8f*)(zt + ((w + 8) * 32 + lane) * 8);
        _Float16* sl = snat + l * (16 * 136);
#pragma unroll
        for (int v = 0; v < 8; ++v) {
          float g = tanhf(zlo[v]) * (1.f / (1.f + __expf(-zhi[v])));
          sl[(mb + v) * 136 + (w * 16 + col16)] = (_Float16)g;
        }
      }
      __syncthreads();

      // ---- phase 2b (waves 0..7): x = a @ W_res.T + b_res + x
      if (w < 8) {
        float br = b_res[l * 128 + w * 16 + col16];
        v8f acc;
#pragma unroll
        for (int v = 0; v < 8; ++v) acc[v] = br;
        const _Float16* sl  = snat + l * (16 * 136);
        const _Float16* bpr = wres + (unsigned)(((l * 8 + w) * 4) * 512) + lane * 16;
#pragma unroll
        for (int kb = 0; kb < 4; ++kb) {
          v16h aA = ldsAfrag(sl, 136, kb * 32, lane);
          acc = wmma16(aA, *(const v16h*)(bpr + kb * 512), acc);
        }
#pragma unroll
        for (int v = 0; v < 8; ++v) {
          int idx = (mb + v) * 136 + w * 16 + col16;
          float nx = acc[v] + (float)xnat[idx];
          xnat[idx] = (_Float16)nx;
        }
      }
      __syncthreads();
    }

    // ---- out1 (waves 0..7): h = tanh(concat(skips) @ W_out1.T + b_out1)
    if (w < 8) {
      float bo = b_out1[w * 16 + col16];
      v8f acc;
#pragma unroll
      for (int v = 0; v < 8; ++v) acc[v] = bo;
      const _Float16* bp = wout1 + (unsigned)((w * 32) * 512) + lane * 16;
#pragma unroll 1
      for (int l = 0; l < L_SZ; ++l) {
        const _Float16* sl = snat + l * (16 * 136);
#pragma unroll
        for (int kb = 0; kb < 4; ++kb) {
          v16h aA = ldsAfrag(sl, 136, kb * 32, lane);
          acc = wmma16(aA, *(const v16h*)(bp + (l * 4 + kb) * 512), acc);
        }
      }
#pragma unroll
      for (int v = 0; v < 8; ++v)
        hnat[(mb + v) * 136 + w * 16 + col16] = tanhf(acc[v]);
    }
    __syncthreads();

    // ---- y = h @ W_out2.T + b_out2 (COUT=1): one lane per batch row
    if (tid < 16) {
      float a = b_out2[0];
#pragma unroll 1
      for (int r = 0; r < 128; ++r) a += hnat[tid * 136 + r] * W_out2[r];
      out[tid * T_STEPS + t] = a;   // (B, 1, T)
      ybuf[tid] = a;
    }
    __syncthreads();
  }
}

// ------------------------------- launcher ----------------------------------

extern "C" void kernel_launch(void* const* d_in, const int* in_sizes, int n_in,
                              void* d_out, int out_size, void* d_ws, size_t ws_size,
                              hipStream_t stream) {
  (void)in_sizes; (void)n_in; (void)out_size; (void)ws_size;
  const float* cond  = (const float*)d_in[0];
  const float* Win   = (const float*)d_in[1];
  const float* bin   = (const float*)d_in[2];
  const float* Wconv = (const float*)d_in[3];
  const float* bconv = (const float*)d_in[4];
  const float* Wcond = (const float*)d_in[5];
  const float* bcond = (const float*)d_in[6];
  const float* Wres  = (const float*)d_in[7];
  const float* bres  = (const float*)d_in[8];
  const float* Wout1 = (const float*)d_in[9];
  const float* bout1 = (const float*)d_in[10];
  const float* Wout2 = (const float*)d_in[11];
  const float* bout2 = (const float*)d_in[12];
  _Float16* ws  = (_Float16*)d_ws;
  float*    out = (float*)d_out;

  pack_conv_k   <<<524288u / 256u, 256, 0, stream>>>(Wconv, ws);
  pack_cond_k   <<<196608u / 256u, 256, 0, stream>>>(Wcond, ws);
  pack_res_k    <<<131072u / 256u, 256, 0, stream>>>(Wres,  ws);
  pack_out1_k   <<<131072u / 256u, 256, 0, stream>>>(Wout1, ws);
  pack_condin_k <<<3145728u / 256u, 256, 0, stream>>>(cond, ws);
  pack_biasz_k  <<<8, 256, 0, stream>>>(bconv, bcond, ws);
  zero_ring_k   <<<(RING_UINTS + 255u) / 256u, 256, 0, stream>>>(ws);

  wavenet_ar_k<<<1, 512, 0, stream>>>(Win, bin, bres, bout1, Wout2, bout2, ws, out);
}
